// csai_53523882443296
// MI455X (gfx1250) — compile-verified, matrix operand order checked by reading, not run on hardware
//
#include <hip/hip_runtime.h>
#include <math.h>

// Problem sizes (fixed by the reference)
#define Bsz 256
#define Tt  48
#define Dd  89
#define Hh  128
#define Cc  64
#define NHh 8
#define T2  96   // 2*T

typedef __attribute__((ext_vector_type(16))) _Float16 v16h;
typedef __attribute__((ext_vector_type(8)))  _Float16 v8h;
typedef __attribute__((ext_vector_type(8)))  float    v8f;
typedef __attribute__((ext_vector_type(4)))  float    v4f;

// ---------------------------------------------------------------------------
// WMMA fragment loaders (CDNA5 wave32 layout, ISA 7.12.2).
// Per lane: row = lane&15, K sub-base kb = (lane>>4)*8.
// half e -> K = (e>>3)*16 + kb + (e&7): two contiguous 8-element runs.
// All loaders are branch-free; K-padding correctness comes from zero-padded
// f16 weights on the B side.
// ---------------------------------------------------------------------------
__device__ __forceinline__ v16h load_a_f32(const float* __restrict__ src,
                                           int ld, int r0, int k0) {
  const int lane = threadIdx.x & 31;
  const int r = r0 + (lane & 15);
  const int kb = (lane >> 4) << 3;
  const float* p = src + (long)r * ld + k0 + kb;
  v4f a0 = *(const v4f*)(p);
  v4f a1 = *(const v4f*)(p + 4);
  v4f b0 = *(const v4f*)(p + 16);
  v4f b1 = *(const v4f*)(p + 20);
  v16h f;
#pragma unroll
  for (int e = 0; e < 4; ++e) {
    f[e]      = (_Float16)a0[e];
    f[4 + e]  = (_Float16)a1[e];
    f[8 + e]  = (_Float16)b0[e];
    f[12 + e] = (_Float16)b1[e];
  }
  return f;
}

__device__ __forceinline__ v16h load_f16(const _Float16* __restrict__ src,
                                         int ld, int r0, int k0) {
  const int lane = threadIdx.x & 31;
  const int r = r0 + (lane & 15);
  const int kb = (lane >> 4) << 3;
  const _Float16* p = src + (long)r * ld + k0 + kb;
  v8h lo = *(const v8h*)(p);
  v8h hi = *(const v8h*)(p + 16);
  v16h f;
#pragma unroll
  for (int e = 0; e < 8; ++e) {
    f[e] = lo[e];
    f[8 + e] = hi[e];
  }
  return f;
}

__device__ __forceinline__ v8f wmma16(v16h a, v16h b, v8f c) {
  return __builtin_amdgcn_wmma_f32_16x16x32_f16(false, a, false, b,
                                                (short)0, c, false, false);
}

__device__ __forceinline__ float sigmf(float x) { return 1.0f / (1.0f + expf(-x)); }

// ---------------------------------------------------------------------------
// Generic WMMA GEMM: out[M,N] = A[M,K] @ W16[N,K]^T + bias (optional GELU).
// Requires M%16==0, N%16==0, K%32==0 (true for every call site) -> no bounds.
// One wave per 16x16 tile, grid-stride.
// ---------------------------------------------------------------------------
__global__ __launch_bounds__(256) void gemm_wmma(const float* __restrict__ A,
                                                 const _Float16* __restrict__ W,
                                                 const float* __restrict__ bias,
                                                 float* __restrict__ out,
                                                 int M, int N, int K, int act) {
  const int nt = N >> 4;
  const int mt = M >> 4;
  const int total = mt * nt;
  const int wave   = blockIdx.x * (blockDim.x >> 5) + (threadIdx.x >> 5);
  const int nwaves = gridDim.x * (blockDim.x >> 5);
  const int lane = threadIdx.x & 31;
  const int nn = lane & 15, mb = (lane >> 4) << 3;
  for (int tidx = wave; tidx < total; tidx += nwaves) {
    const int tm = tidx / nt, tn = tidx - tm * nt;
    const int m0 = tm << 4, n0 = tn << 4;
    v8f acc = {};
    for (int k0 = 0; k0 < K; k0 += 32) {
      v16h a = load_a_f32(A, K, m0, k0);
      v16h b = load_f16(W, K, n0, k0);
      acc = wmma16(a, b, acc);
    }
    const float bv = bias[n0 + nn];
#pragma unroll
    for (int i = 0; i < 8; ++i) {
      float v = acc[i] + bv;
      if (act == 1) v = 0.5f * v * (1.0f + erff(v * 0.70710678118654752f));
      out[(long)(m0 + mb + i) * N + n0 + nn] = v;
    }
  }
}

// ---------------------------------------------------------------------------
// Pad + convert f32 weight [nri,ldi(cols nki)] -> f16 [nro,ldo], zero padded.
// mask_diag: zero the r==c diagonal (FeatureRegression mask).
// ---------------------------------------------------------------------------
__global__ void pad_f16_kernel(const float* __restrict__ in,
                               _Float16* __restrict__ out, int nro, int ldo,
                               int nri, int nki, int ldi, int mask_diag) {
  int idx = blockIdx.x * blockDim.x + threadIdx.x;
  if (idx >= nro * ldo) return;
  int r = idx / ldo, c = idx - r * ldo;
  float v = 0.0f;
  if (r < nri && c < nki) v = in[(long)r * ldi + c];
  if (mask_diag && r == c) v = 0.0f;
  out[idx] = (_Float16)v;
}

// Positional encoding table pe[T][C].
__global__ void pe_kernel(float* __restrict__ pe) {
  int tid = blockIdx.x * blockDim.x + threadIdx.x;
  if (tid >= Tt * Cc) return;
  int t = tid / Cc, c = tid - t * Cc;
  int j = c >> 1;
  float div = expf(-(float)(2 * j) * (logf(10000.0f) / (float)Cc));
  float a = (float)t * div;
  pe[tid] = (c & 1) ? cosf(a) : sinf(a);
}

// ---------------------------------------------------------------------------
// Decay_obs + time encoder: one block per (b,t).
// ---------------------------------------------------------------------------
__global__ __launch_bounds__(128) void encode_kernel(
    const float* __restrict__ deltas, const float* __restrict__ medians,
    const float* __restrict__ last_obs, const float* __restrict__ W_wo,
    const float* __restrict__ b_wo, const float* __restrict__ W_inp,
    const float* __restrict__ b_inp, const float* __restrict__ pe,
    float* __restrict__ data) {
  __shared__ float dd[Dd], df[Dd], lo[Dd];
  const int bt = blockIdx.x;
  const int b = bt / Tt, t = bt - b * Tt;
  const int tid = threadIdx.x;
  if (tid < Dd) {
    float v = deltas[(long)bt * Dd + tid] - medians[tid];
    dd[tid] = v;
    lo[tid] = last_obs[(long)bt * Dd + tid];
  }
  __syncthreads();
  if (tid < Dd) {
    float s = 0.0f;
    for (int d = 0; d < Dd; ++d) s += dd[d] * W_wo[tid * Dd + d];
    s += b_wo[tid];
    float x = dd[tid];
    float sg = (x > 0.0f) ? 1.0f : ((x < 0.0f) ? -1.0f : 0.0f);
    df[tid] = 0.5f * (1.0f - tanhf(sg * fabsf(s)));
  }
  __syncthreads();
  if (tid < Cc) {
    float s1 = 0.0f, s2 = 0.0f;
    for (int d = 0; d < Dd; ++d) {
      float w = W_inp[tid * Dd + d];
      s1 += lo[d] * w;
      s2 += df[d] * w;
    }
    float base = b_inp[tid] + pe[t * Cc + tid];
    data[((long)b * T2 + t) * Cc + tid]      = s1 + base;
    data[((long)b * T2 + Tt + t) * Cc + tid] = s2 + base;
  }
}

// ---------------------------------------------------------------------------
// Fused attention over batch axis (seq len = 256), one block per (n,h).
// HD=8 -> VALU flash softmax (a K=32 WMMA would waste 75% of its MACs).
// ---------------------------------------------------------------------------
__global__ __launch_bounds__(256) void attn_kernel(const float* __restrict__ qkv,
                                                   float* __restrict__ o) {
  __shared__ float Qs[Bsz][9], Ks[Bsz][9], Vs[Bsz][9];
  const int nh = blockIdx.x;
  const int n = nh >> 3, h = nh & 7;
  const int tid = threadIdx.x;
  {
    long base = ((long)tid * T2 + n) * (3 * Cc) + h * 8;
#pragma unroll
    for (int e = 0; e < 8; ++e) {
      Qs[tid][e] = qkv[base + e];
      Ks[tid][e] = qkv[base + Cc + e];
      Vs[tid][e] = qkv[base + 2 * Cc + e];
    }
  }
  __syncthreads();
  const float scale = 0.35355339059327373f;  // 1/sqrt(8)
  float qr[8];
#pragma unroll
  for (int e = 0; e < 8; ++e) qr[e] = Qs[tid][e];
  float m = -1e30f, l = 0.0f;
  float acc[8] = {0, 0, 0, 0, 0, 0, 0, 0};
  for (int tt = 0; tt < Bsz; ++tt) {
    float d = 0.0f;
#pragma unroll
    for (int e = 0; e < 8; ++e) d += qr[e] * Ks[tt][e];
    d *= scale;
    float nm = fmaxf(m, d);
    float corr = expf(m - nm);
    float p = expf(d - nm);
    l = l * corr + p;
#pragma unroll
    for (int e = 0; e < 8; ++e) acc[e] = acc[e] * corr + p * Vs[tt][e];
    m = nm;
  }
  float inv = 1.0f / l;
  long ob = ((long)tid * T2 + n) * Cc + h * 8;
#pragma unroll
  for (int e = 0; e < 8; ++e) o[ob + e] = acc[e] * inv;
}

// out = LayerNorm(a + b) over C=64, thread per row.
__global__ __launch_bounds__(256) void add_ln_kernel(
    const float* __restrict__ a, const float* __restrict__ b,
    const float* __restrict__ g, const float* __restrict__ be,
    float* __restrict__ out, int rows) {
  int r = blockIdx.x * blockDim.x + threadIdx.x;
  if (r >= rows) return;
  long base = (long)r * Cc;
  float mean = 0.0f;
  for (int c = 0; c < Cc; ++c) mean += a[base + c] + b[base + c];
  mean *= (1.0f / Cc);
  float var = 0.0f;
  for (int c = 0; c < Cc; ++c) {
    float v = a[base + c] + b[base + c] - mean;
    var += v * v;
  }
  var *= (1.0f / Cc);
  float inv = rsqrtf(var + 1e-5f);
  for (int c = 0; c < Cc; ++c) {
    float v = a[base + c] + b[base + c];
    out[base + c] = (v - mean) * inv * g[c] + be[c];
  }
}

// h0[b,j] = sum_n hp[b,n,j] * W_op2[n] + b_op2[0]
__global__ __launch_bounds__(256) void h0_kernel(const float* __restrict__ hp,
                                                 const float* __restrict__ W_op2,
                                                 const float* __restrict__ b_op2,
                                                 float* __restrict__ h0) {
  int idx = blockIdx.x * blockDim.x + threadIdx.x;
  if (idx >= Bsz * Hh) return;
  int b = idx >> 7, j = idx & 127;
  float s = 0.0f;
  for (int n = 0; n < T2; ++n) s += hp[((long)b * T2 + n) * Hh + j] * W_op2[n];
  h0[idx] = s + b_op2[0];
}

// ---------------------------------------------------------------------------
// Fused 48-step GRU-D scan. 16 WGs x 16 batch rows. Six WMMA GEMMs per step.
// Weights are zero-padded f16 (K: 89->96, 178->192) so A-side padding lanes
// multiply zero -> no bounds checks on fragment loads.
// LDS:
//   sh_h   f32 [16][128]  8 KB
//   sh_cat f16 [16][192]  6 KB   ([x|m] concat operand)
//   reg    f32 [12288]    48 KB:
//     xh @0, xu @1536, beta @3072 (16x96 each)
//     rn @4608, rd @4864 (reduction)
//     db (f16 [16][96]) @5120  (staged d_t for step 1)
//     gi @0 (16x384), gh @6144 (16x384)  -- overlay, live only steps 8-10
// ---------------------------------------------------------------------------
#define ROWS 16
#define LDX 4272  // T*D row stride

__global__ __launch_bounds__(256) void scan_kernel(
    const float* __restrict__ x, const float* __restrict__ mask,
    const float* __restrict__ deltas, const _Float16* __restrict__ Wdh16,
    const float* __restrict__ b_dh, const float* __restrict__ W_dx,
    const float* __restrict__ b_dx, const _Float16* __restrict__ Whist16,
    const float* __restrict__ b_hist, const _Float16* __restrict__ Wfeat16,
    const float* __restrict__ b_feat, const _Float16* __restrict__ Wwc16,
    const float* __restrict__ b_wc, const _Float16* __restrict__ Wih16,
    const float* __restrict__ b_ih, const _Float16* __restrict__ Whh16,
    const float* __restrict__ b_hh, const float* __restrict__ W_cls,
    const float* __restrict__ b_cls, const float* __restrict__ h0,
    float* __restrict__ x_imp_out, float* __restrict__ hid_out,
    float* __restrict__ y_out, float* __restrict__ y_score,
    float* __restrict__ loss_num, float* __restrict__ loss_den) {
  __shared__ __attribute__((aligned(16))) float sh_h[ROWS * Hh];
  __shared__ __attribute__((aligned(16))) _Float16 sh_cat[ROWS * 192];
  __shared__ __attribute__((aligned(16))) float reg[12288];
  float* sh_xh   = reg;
  float* sh_xu   = reg + 1536;
  float* sh_beta = reg + 3072;
  float* rn      = reg + 4608;
  float* rd      = reg + 4864;
  _Float16* db   = (_Float16*)(reg + 5120);  // [16][96]
  float* sh_gi   = reg;                      // [16][384]
  float* sh_gh   = reg + 6144;               // [16][384]

  const int tid = threadIdx.x;
  const int wv  = tid >> 5;
  const int lane = tid & 31;
  const int nn = lane & 15, mb = (lane >> 4) << 3;
  const int b0 = blockIdx.x * ROWS;

  for (int i = tid; i < ROWS * Hh; i += 256)
    sh_h[i] = h0[(long)(b0 + i / Hh) * Hh + (i % Hh)];
  for (int i = tid; i < ROWS * 192; i += 256) sh_cat[i] = (_Float16)0.0f;
  __syncthreads();

  for (int t = 0; t < Tt; ++t) {
    const float* dt0 = deltas + (long)b0 * LDX + t * Dd;
    const float* xt0 = x      + (long)b0 * LDX + t * Dd;
    const float* mt0 = mask   + (long)b0 * LDX + t * Dd;

    // stage d_t -> f16 padded [16][96]
    for (int idx = tid; idx < ROWS * 96; idx += 256) {
      int r = idx / 96, c = idx - r * 96;
      db[idx] = (c < Dd) ? (_Float16)dt0[(long)r * LDX + c] : (_Float16)0.0f;
    }
    __syncthreads();

    // 1) gamma_h = exp(-relu(d_t @ W_dh^T + b_dh));  h *= gamma_h
    {
      int n0 = wv << 4;  // 8 waves cover N=128
      v8f acc = {};
      for (int k0 = 0; k0 < 96; k0 += 32) {
        v16h a = load_f16(db, 96, 0, k0);
        v16h b = load_f16(Wdh16, 96, n0, k0);
        acc = wmma16(a, b, acc);
      }
#pragma unroll
      for (int i = 0; i < 8; ++i) {
        float v = acc[i] + b_dh[n0 + nn];
        sh_h[(mb + i) * Hh + n0 + nn] *= expf(-fmaxf(v, 0.0f));
      }
    }
    __syncthreads();

    // 2) x_h = h @ W_hist^T + b_hist
    if (wv < 6) {
      int n0 = wv << 4;
      v8f acc = {};
      for (int k0 = 0; k0 < Hh; k0 += 32) {
        v16h a = load_a_f32(sh_h, Hh, 0, k0);
        v16h b = load_f16(Whist16, Hh, n0, k0);
        acc = wmma16(a, b, acc);
      }
#pragma unroll
      for (int i = 0; i < 8; ++i) {
        int c = n0 + nn;
        if (c < Dd) sh_xh[(mb + i) * 96 + c] = acc[i] + b_hist[c];
      }
    }
    __syncthreads();

    // 3) x_r = m*x + (1-m)*x_h -> cat[:,0:89] (f16)
    for (int idx = tid; idx < ROWS * Dd; idx += 256) {
      int r = idx / Dd, d = idx - r * Dd;
      float m = mt0[(long)r * LDX + d];
      float xv = xt0[(long)r * LDX + d];
      sh_cat[r * 192 + d] = (_Float16)(m * xv + (1.0f - m) * sh_xh[r * 96 + d]);
    }
    __syncthreads();

    // 4) xu = x_r @ (W_feat masked)^T + b_feat
    if (wv < 6) {
      int n0 = wv << 4;
      v8f acc = {};
      for (int k0 = 0; k0 < 96; k0 += 32) {
        v16h a = load_f16(sh_cat, 192, 0, k0);
        v16h b = load_f16(Wfeat16, 96, n0, k0);
        acc = wmma16(a, b, acc);
      }
#pragma unroll
      for (int i = 0; i < 8; ++i) {
        int c = n0 + nn;
        if (c < Dd) sh_xu[(mb + i) * 96 + c] = acc[i] + b_feat[c];
      }
    }
    __syncthreads();

    // 5) gamma_x; cat = [gamma_x | m]
    for (int idx = tid; idx < ROWS * Dd; idx += 256) {
      int r = idx / Dd, d = idx - r * Dd;
      float dv = dt0[(long)r * LDX + d];
      float gx = expf(-fmaxf(dv * W_dx[d * Dd + d] + b_dx[d], 0.0f));
      sh_cat[r * 192 + d] = (_Float16)gx;
      sh_cat[r * 192 + Dd + d] = (_Float16)mt0[(long)r * LDX + d];
    }
    __syncthreads();

    // 6) beta = cat @ W_wc^T + b_wc   (K padded 178->192)
    if (wv < 6) {
      int n0 = wv << 4;
      v8f acc = {};
      for (int k0 = 0; k0 < 192; k0 += 32) {
        v16h a = load_f16(sh_cat, 192, 0, k0);
        v16h b = load_f16(Wwc16, 192, n0, k0);
        acc = wmma16(a, b, acc);
      }
#pragma unroll
      for (int i = 0; i < 8; ++i) {
        int c = n0 + nn;
        if (c < Dd) sh_beta[(mb + i) * 96 + c] = acc[i] + b_wc[c];
      }
    }
    __syncthreads();

    // 7) x_comb, loss partials, x_imp; cat = [x_imp | m]
    float lnum = 0.0f, lden = 0.0f;
    for (int idx = tid; idx < ROWS * Dd; idx += 256) {
      int r = idx / Dd, d = idx - r * Dd;
      float be = sh_beta[r * 96 + d];
      float xc = be * sh_xu[r * 96 + d] + (1.0f - be) * sh_xh[r * 96 + d];
      float m = mt0[(long)r * LDX + d];
      float xv = xt0[(long)r * LDX + d];
      lnum += fabsf(xv - xc) * m;
      lden += m;
      float xi = m * xv + (1.0f - m) * xc;
      x_imp_out[(long)(b0 + r) * LDX + t * Dd + d] = xi;
      sh_cat[r * 192 + d] = (_Float16)xi;
      sh_cat[r * 192 + Dd + d] = (_Float16)m;
    }
    rn[tid] = lnum;
    rd[tid] = lden;
    __syncthreads();
    for (int s = 128; s > 0; s >>= 1) {
      if (tid < s) { rn[tid] += rn[tid + s]; rd[tid] += rd[tid + s]; }
      __syncthreads();
    }
    if (tid == 0) {
      loss_num[blockIdx.x * Tt + t] = rn[0];
      loss_den[blockIdx.x * Tt + t] = rd[0];
    }
    __syncthreads();

    // 8) gi = [x_imp|m] @ W_ih^T + b_ih   (N=384, K=192)
    for (int tn = wv; tn < 24; tn += 8) {
      int n0 = tn << 4;
      v8f acc = {};
      for (int k0 = 0; k0 < 192; k0 += 32) {
        v16h a = load_f16(sh_cat, 192, 0, k0);
        v16h b = load_f16(Wih16, 192, n0, k0);
        acc = wmma16(a, b, acc);
      }
#pragma unroll
      for (int i = 0; i < 8; ++i)
        sh_gi[(mb + i) * 384 + n0 + nn] = acc[i] + b_ih[n0 + nn];
    }
    __syncthreads();

    // 9) gh = h @ W_hh^T + b_hh   (N=384, K=128)
    for (int tn = wv; tn < 24; tn += 8) {
      int n0 = tn << 4;
      v8f acc = {};
      for (int k0 = 0; k0 < Hh; k0 += 32) {
        v16h a = load_a_f32(sh_h, Hh, 0, k0);
        v16h b = load_f16(Whh16, Hh, n0, k0);
        acc = wmma16(a, b, acc);
      }
#pragma unroll
      for (int i = 0; i < 8; ++i)
        sh_gh[(mb + i) * 384 + n0 + nn] = acc[i] + b_hh[n0 + nn];
    }
    __syncthreads();

    // 10) GRU update
    for (int idx = tid; idx < ROWS * Hh; idx += 256) {
      int r = idx / Hh, j = idx - r * Hh;
      float ir = sh_gi[r * 384 + j], iz = sh_gi[r * 384 + 128 + j],
            ig = sh_gi[r * 384 + 256 + j];
      float hr = sh_gh[r * 384 + j], hz = sh_gh[r * 384 + 128 + j],
            hg = sh_gh[r * 384 + 256 + j];
      float rr = sigmf(ir + hr);
      float z  = sigmf(iz + hz);
      float g  = tanhf(ig + rr * hg);
      float hn = (1.0f - z) * g + z * sh_h[r * Hh + j];
      sh_h[r * Hh + j] = hn;
      hid_out[(long)(b0 + r) * (Tt * Hh) + t * Hh + j] = hn;
    }
    __syncthreads();
  }

  // classifier head
  for (int r = tid; r < ROWS; r += 256) {
    float s = 0.0f;
    for (int j = 0; j < Hh; ++j) s += sh_h[r * Hh + j] * W_cls[j];
    s += b_cls[0];
    y_out[b0 + r] = s;
    y_score[b0 + r] = sigmf(s);
  }
}

// loss = sum_t ( sum_wg num ) / ( sum_wg den + 1e-5 )
__global__ void loss_kernel(const float* __restrict__ num,
                            const float* __restrict__ den,
                            float* __restrict__ out) {
  __shared__ float acc[Tt];
  int tid = threadIdx.x;
  if (tid < Tt) {
    float n = 0.0f, d = 0.0f;
    for (int w = 0; w < 16; ++w) { n += num[w * Tt + tid]; d += den[w * Tt + tid]; }
    acc[tid] = n / (d + 1e-5f);
  }
  __syncthreads();
  if (tid == 0) {
    float s = 0.0f;
    for (int t = 0; t < Tt; ++t) s += acc[t];
    out[0] = s;
  }
}

// ---------------------------------------------------------------------------
extern "C" void kernel_launch(void* const* d_in, const int* in_sizes, int n_in,
                              void* d_out, int out_size, void* d_ws,
                              size_t ws_size, hipStream_t stream) {
  (void)in_sizes; (void)n_in; (void)out_size; (void)ws_size;
  const float* x        = (const float*)d_in[0];
  const float* mask     = (const float*)d_in[1];
  const float* deltas   = (const float*)d_in[2];
  const float* last_obs = (const float*)d_in[3];
  const float* medians  = (const float*)d_in[4];
  const float* W_dh  = (const float*)d_in[5];   const float* b_dh  = (const float*)d_in[6];
  const float* W_dx  = (const float*)d_in[7];   const float* b_dx  = (const float*)d_in[8];
  const float* W_hist= (const float*)d_in[9];   const float* b_hist= (const float*)d_in[10];
  const float* W_feat= (const float*)d_in[11];  const float* b_feat= (const float*)d_in[12];
  const float* W_wc  = (const float*)d_in[13];  const float* b_wc  = (const float*)d_in[14];
  const float* W_wo  = (const float*)d_in[15];  const float* b_wo  = (const float*)d_in[16];
  const float* W_cls = (const float*)d_in[17];  const float* b_cls = (const float*)d_in[18];
  const float* W_ih  = (const float*)d_in[19];  const float* W_hh  = (const float*)d_in[20];
  const float* b_ih  = (const float*)d_in[21];  const float* b_hh  = (const float*)d_in[22];
  const float* W_inp = (const float*)d_in[23];  const float* b_inp = (const float*)d_in[24];
  const float* W_op1 = (const float*)d_in[25];  const float* b_op1 = (const float*)d_in[26];
  const float* W_op2 = (const float*)d_in[27];  const float* b_op2 = (const float*)d_in[28];
  const float* attn_in_w  = (const float*)d_in[29]; const float* attn_in_b  = (const float*)d_in[30];
  const float* attn_out_w = (const float*)d_in[31]; const float* attn_out_b = (const float*)d_in[32];
  const float* ln1_g = (const float*)d_in[33];  const float* ln1_b = (const float*)d_in[34];
  const float* ln2_g = (const float*)d_in[35];  const float* ln2_b = (const float*)d_in[36];
  const float* ff1_w = (const float*)d_in[37];  const float* ff1_b = (const float*)d_in[38];
  const float* ff2_w = (const float*)d_in[39];  const float* ff2_b = (const float*)d_in[40];

  // outputs: x_imp [B,T,D] | x_loss [1] | hiddens [B,T,H] | y_out [B] | y_score [B]
  float* out = (float*)d_out;
  float* o_ximp = out;
  float* o_loss = out + (long)Bsz * Tt * Dd;
  float* o_hid  = o_loss + 1;
  float* o_y    = o_hid + (long)Bsz * Tt * Hh;
  float* o_ys   = o_y + Bsz;

  // workspace: f32 region then f16 region
  float* ws = (float*)d_ws;
  size_t off = 0;
  float* data  = ws + off; off += (size_t)Bsz * T2 * Cc;        // 1,572,864
  float* qkv   = ws + off; off += (size_t)Bsz * T2 * 3 * Cc;    // (reused: proj, ff2out)
  float* atno  = ws + off; off += (size_t)Bsz * T2 * Cc;        // (reused: ff1out)
  float* x1    = ws + off; off += (size_t)Bsz * T2 * Cc;
  float* x2    = ws + off; off += (size_t)Bsz * T2 * Cc;
  float* hp    = ws + off; off += (size_t)Bsz * T2 * Hh;
  float* h0buf = ws + off; off += (size_t)Bsz * Hh;
  float* pe    = ws + off; off += (size_t)Tt * Cc;
  float* lnum  = ws + off; off += 16 * Tt;
  float* lden  = ws + off; off += 16 * Tt;

  _Float16* wsH = (_Float16*)(ws + off);
  size_t hoff = 0;
  _Float16* Wdh16   = wsH + hoff; hoff += 128 * 96;
  _Float16* Whist16 = wsH + hoff; hoff += 96 * 128;
  _Float16* Wfeat16 = wsH + hoff; hoff += 96 * 96;
  _Float16* Wwc16   = wsH + hoff; hoff += 96 * 192;
  _Float16* Wih16   = wsH + hoff; hoff += 384 * 192;
  _Float16* Whh16   = wsH + hoff; hoff += 384 * 128;
  _Float16* AttnIn16  = wsH + hoff; hoff += 192 * 64;
  _Float16* AttnOut16 = wsH + hoff; hoff += 64 * 64;
  _Float16* Ff1_16    = wsH + hoff; hoff += 64 * 64;
  _Float16* Ff2_16    = wsH + hoff; hoff += 64 * 64;
  _Float16* Wop1_16   = wsH + hoff; hoff += 128 * 64;

  const int Mrows = Bsz * T2;  // 24576
  auto gemm_blocks = [](int M, int N) {
    int tiles = (M / 16) * (N / 16);
    return (tiles + 7) / 8;
  };
  auto pad_blocks = [](int n) { return (n + 255) / 256; };

  // weight conversion / padding
  pad_f16_kernel<<<pad_blocks(128 * 96), 256, 0, stream>>>(W_dh, Wdh16, 128, 96, 128, 89, 89, 0);
  pad_f16_kernel<<<pad_blocks(96 * 128), 256, 0, stream>>>(W_hist, Whist16, 96, 128, 89, 128, 128, 0);
  pad_f16_kernel<<<pad_blocks(96 * 96), 256, 0, stream>>>(W_feat, Wfeat16, 96, 96, 89, 89, 89, 1);
  pad_f16_kernel<<<pad_blocks(96 * 192), 256, 0, stream>>>(W_wc, Wwc16, 96, 192, 89, 178, 178, 0);
  pad_f16_kernel<<<pad_blocks(384 * 192), 256, 0, stream>>>(W_ih, Wih16, 384, 192, 384, 178, 178, 0);
  pad_f16_kernel<<<pad_blocks(384 * 128), 256, 0, stream>>>(W_hh, Whh16, 384, 128, 384, 128, 128, 0);
  pad_f16_kernel<<<pad_blocks(192 * 64), 256, 0, stream>>>(attn_in_w, AttnIn16, 192, 64, 192, 64, 64, 0);
  pad_f16_kernel<<<pad_blocks(64 * 64), 256, 0, stream>>>(attn_out_w, AttnOut16, 64, 64, 64, 64, 64, 0);
  pad_f16_kernel<<<pad_blocks(64 * 64), 256, 0, stream>>>(ff1_w, Ff1_16, 64, 64, 64, 64, 64, 0);
  pad_f16_kernel<<<pad_blocks(64 * 64), 256, 0, stream>>>(ff2_w, Ff2_16, 64, 64, 64, 64, 64, 0);
  pad_f16_kernel<<<pad_blocks(128 * 64), 256, 0, stream>>>(W_op1, Wop1_16, 128, 64, 128, 64, 64, 0);
  pe_kernel<<<pad_blocks(Tt * Cc), 256, 0, stream>>>(pe);

  encode_kernel<<<Bsz * Tt, 128, 0, stream>>>(deltas, medians, last_obs, W_wo,
                                              b_wo, W_inp, b_inp, pe, data);
  // qkv = data @ attn_in_w^T + attn_in_b
  gemm_wmma<<<gemm_blocks(Mrows, 192), 256, 0, stream>>>(data, AttnIn16,
      attn_in_b, qkv, Mrows, 192, Cc, 0);
  attn_kernel<<<T2 * NHh, 256, 0, stream>>>(qkv, atno);
  // proj = attn_o @ attn_out_w^T + attn_out_b  (reuse qkv buffer)
  gemm_wmma<<<gemm_blocks(Mrows, Cc), 256, 0, stream>>>(atno, AttnOut16,
      attn_out_b, qkv, Mrows, Cc, Cc, 0);
  add_ln_kernel<<<(Mrows + 255) / 256, 256, 0, stream>>>(data, qkv, ln1_g,
      ln1_b, x1, Mrows);
  // ff1out = gelu(x1 @ ff1_w^T + ff1_b)  (into atno)
  gemm_wmma<<<gemm_blocks(Mrows, Cc), 256, 0, stream>>>(x1, Ff1_16, ff1_b,
      atno, Mrows, Cc, Cc, 1);
  // ff2out = ff1out @ ff2_w^T + ff2_b  (into qkv)
  gemm_wmma<<<gemm_blocks(Mrows, Cc), 256, 0, stream>>>(atno, Ff2_16, ff2_b,
      qkv, Mrows, Cc, Cc, 0);
  add_ln_kernel<<<(Mrows + 255) / 256, 256, 0, stream>>>(x1, qkv, ln2_g,
      ln2_b, x2, Mrows);
  // hp = x2 @ W_op1^T + b_op1
  gemm_wmma<<<gemm_blocks(Mrows, Hh), 256, 0, stream>>>(x2, Wop1_16, b_op1, hp,
      Mrows, Hh, Cc, 0);
  h0_kernel<<<(Bsz * Hh + 255) / 256, 256, 0, stream>>>(hp, W_op2, b_op2, h0buf);
  scan_kernel<<<Bsz / ROWS, 256, 0, stream>>>(
      x, mask, deltas, Wdh16, b_dh, W_dx, b_dx, Whist16, b_hist, Wfeat16,
      b_feat, Wwc16, b_wc, Wih16, b_ih, Whh16, b_hh, W_cls, b_cls, h0buf,
      o_ximp, o_hid, o_y, o_ys, lnum, lden);
  loss_kernel<<<1, 64, 0, stream>>>(lnum, lden, o_loss);
}
